// VisionTransformer_78194174591514
// MI455X (gfx1250) — compile-verified, matrix operand order checked by reading, not run on hardware
//
#include <hip/hip_runtime.h>
#include <hip/hip_bf16.h>
#include <math.h>

typedef __attribute__((ext_vector_type(16))) _Float16 v16h;
typedef __attribute__((ext_vector_type(4)))  _Float16 h4;
typedef __attribute__((ext_vector_type(8)))  float    v8f;

// ---------------- problem constants ----------------
constexpr int cD   = 512;
constexpr int cH   = 8;
constexpr int cDH  = 64;
constexpr int cL   = 8;
constexpr int cSEQ = 197;
constexpr int cFF  = 2048;
constexpr int cB   = 64;
constexpr int cNP  = 196;   // patches per image
constexpr int cPK  = 768;   // 3*16*16 patch K
constexpr int cSP  = 208;   // padded score row stride
constexpr float cEPS = 1e-5f;

// ---------------- GEMM tiling ----------------
constexpr int BM = 128, BN = 64, BK = 32;
constexpr int LDT = BK + 8;          // 40-half pitch -> conflict-free frag reads

__device__ __forceinline__ float gelu_exact(float x) {
    return 0.5f * x * (1.0f + erff(x * 0.70710678118654752440f));
}

// ---- staging helpers: global(f32) -> regs -> LDS(f16) ----
// A tile: BM x BK = 4096 elems, 16 per thread (4 x float4 chunks)
__device__ __forceinline__
void load_a_tile(const float* __restrict__ A, int lda, int M, int K,
                 int rowBase, int k0, int tid, bool rowsIn, float* aReg)
{
    if (rowsIn && (k0 + BK) <= K) {
        #pragma unroll
        for (int i = 0; i < 4; ++i) {
            int e  = (i * 256 + tid) * 4;
            int rr = e >> 5, cc = e & 31;
            const float4 v = *(const float4*)(A + (size_t)(rowBase + rr) * lda + (k0 + cc));
            aReg[i*4+0] = v.x; aReg[i*4+1] = v.y; aReg[i*4+2] = v.z; aReg[i*4+3] = v.w;
        }
    } else {
        #pragma unroll
        for (int i = 0; i < 4; ++i) {
            int e  = (i * 256 + tid) * 4;
            int rr = e >> 5, cc = e & 31;
            int gr = rowBase + rr;
            int grc = gr < M ? gr : (M - 1);
            #pragma unroll
            for (int s = 0; s < 4; ++s) {
                int gc  = k0 + cc + s;
                int gcc = gc < K ? gc : (K - 1);
                float v = A[(size_t)grc * lda + gcc];   // clamped, always safe
                aReg[i*4+s] = (gr < M && gc < K) ? v : 0.0f;
            }
        }
    }
}

__device__ __forceinline__
void store_a_tile(const float* aReg, _Float16* buf, int tid)
{
    #pragma unroll
    for (int i = 0; i < 4; ++i) {
        int e  = (i * 256 + tid) * 4;
        int rr = e >> 5, cc = e & 31;
        h4 hv;
        hv[0] = (_Float16)aReg[i*4+0]; hv[1] = (_Float16)aReg[i*4+1];
        hv[2] = (_Float16)aReg[i*4+2]; hv[3] = (_Float16)aReg[i*4+3];
        *(h4*)(buf + rr * LDT + cc) = hv;   // 8B-aligned ds_store_b64
    }
}

// W tile (wTrans==0): W is [N,K]; BN x BK = 2048 elems, 8 per thread
__device__ __forceinline__
void load_w_tile(const float* __restrict__ W, int ldw, int N, int K,
                 int colBase, int k0, int tid, bool colsIn, float* wReg)
{
    if (colsIn && (k0 + BK) <= K) {
        #pragma unroll
        for (int i = 0; i < 2; ++i) {
            int e  = (i * 256 + tid) * 4;
            int nn = e >> 5, cc = e & 31;
            const float4 v = *(const float4*)(W + (size_t)(colBase + nn) * ldw + (k0 + cc));
            wReg[i*4+0] = v.x; wReg[i*4+1] = v.y; wReg[i*4+2] = v.z; wReg[i*4+3] = v.w;
        }
    } else {
        #pragma unroll
        for (int i = 0; i < 2; ++i) {
            int e  = (i * 256 + tid) * 4;
            int nn = e >> 5, cc = e & 31;
            int gn = colBase + nn;
            int gnc = gn < N ? gn : (N - 1);
            #pragma unroll
            for (int s = 0; s < 4; ++s) {
                int gc  = k0 + cc + s;
                int gcc = gc < K ? gc : (K - 1);
                float v = W[(size_t)gnc * ldw + gcc];
                wReg[i*4+s] = (gn < N && gc < K) ? v : 0.0f;
            }
        }
    }
}

__device__ __forceinline__
void store_w_tile(const float* wReg, _Float16* buf, int tid)
{
    #pragma unroll
    for (int i = 0; i < 2; ++i) {
        int e  = (i * 256 + tid) * 4;
        int nn = e >> 5, cc = e & 31;
        h4 hv;
        hv[0] = (_Float16)wReg[i*4+0]; hv[1] = (_Float16)wReg[i*4+1];
        hv[2] = (_Float16)wReg[i*4+2]; hv[3] = (_Float16)wReg[i*4+3];
        *(h4*)(buf + nn * LDT + cc) = hv;
    }
}

// W tile (wTrans==1): W is [K,N]; load contiguous along N, scatter into [n][k] LDS
__device__ __forceinline__
void load_w_tile_t(const float* __restrict__ W, int ldw, int N, int K,
                   int colBase, int k0, int tid, bool colsIn, float* wReg)
{
    if (colsIn && (k0 + BK) <= K) {
        #pragma unroll
        for (int i = 0; i < 2; ++i) {
            int e  = (i * 256 + tid) * 4;
            int nn = e & 63, cc = e >> 6;
            const float4 v = *(const float4*)(W + (size_t)(k0 + cc) * ldw + (colBase + nn));
            wReg[i*4+0] = v.x; wReg[i*4+1] = v.y; wReg[i*4+2] = v.z; wReg[i*4+3] = v.w;
        }
    } else {
        #pragma unroll
        for (int i = 0; i < 2; ++i) {
            int e  = (i * 256 + tid) * 4;
            int nn = e & 63, cc = e >> 6;
            int gc  = k0 + cc;
            int gcc = gc < K ? gc : (K - 1);
            #pragma unroll
            for (int s = 0; s < 4; ++s) {
                int gn  = colBase + nn + s;
                int gnc = gn < N ? gn : (N - 1);
                float v = W[(size_t)gcc * ldw + gnc];
                wReg[i*4+s] = (gn < N && gc < K) ? v : 0.0f;
            }
        }
    }
}

__device__ __forceinline__
void store_w_tile_t(const float* wReg, _Float16* buf, int tid)
{
    #pragma unroll
    for (int i = 0; i < 2; ++i) {
        int e  = (i * 256 + tid) * 4;
        int nn = e & 63, cc = e >> 6;
        #pragma unroll
        for (int s = 0; s < 4; ++s)
            buf[(nn + s) * LDT + cc] = (_Float16)wReg[i*4+s];
    }
}

// C[M,N] = act( scale * A[M,K] @ W^T + bias ) (+ residual), batched over gridDim.z.
// wTrans==0: W is [N,K] (ld=ldw). wTrans==1: W is [K,N] (ld=ldw).
__global__ __launch_bounds__(256)
void gemm_wmma_kernel(const float* __restrict__ A, size_t aSB, size_t aSH, int lda,
                      const float* __restrict__ W, size_t wSB, size_t wSH, int ldw, int wTrans,
                      const float* __restrict__ bias,
                      const float* __restrict__ resid,
                      float* __restrict__ C, size_t cSB, size_t cSH, int ldc,
                      int M, int N, int K, int batchH, float scale, int act)
{
    __shared__ _Float16 AsBuf[2][BM * LDT];   // 2 x 10240 B
    __shared__ _Float16 WsBuf[2][BN * LDT];   // 2 x  5120 B

    const int zb = blockIdx.z / batchH;
    const int zh = blockIdx.z % batchH;
    A += (size_t)zb * aSB + (size_t)zh * aSH;
    W += (size_t)zb * wSB + (size_t)zh * wSH;
    C += (size_t)zb * cSB + (size_t)zh * cSH;
    if (resid) resid += (size_t)zb * cSB + (size_t)zh * cSH;

    const int rowBase = blockIdx.y * BM;
    const int colBase = blockIdx.x * BN;
    const int tid  = threadIdx.x;
    const int lane = tid & 31;
    const int wv   = tid >> 5;          // wave id 0..7, owns rows [wv*16, wv*16+16)
    const int r    = lane & 15;
    const int hf   = lane >> 4;         // half-wave selects K groups

    const bool rowsIn = (rowBase + BM) <= M;
    const bool colsIn = (colBase + BN) <= N;
    const int  nk     = (K + BK - 1) / BK;

    v8f acc[4];
    #pragma unroll
    for (int nt = 0; nt < 4; ++nt)
        #pragma unroll
        for (int i = 0; i < 8; ++i) acc[nt][i] = 0.0f;

    float aReg[16];
    float wReg[8];

    // prologue: stage tile 0
    load_a_tile(A, lda, M, K, rowBase, 0, tid, rowsIn, aReg);
    if (!wTrans) load_w_tile (W, ldw, N, K, colBase, 0, tid, colsIn, wReg);
    else         load_w_tile_t(W, ldw, N, K, colBase, 0, tid, colsIn, wReg);
    store_a_tile(aReg, AsBuf[0], tid);
    if (!wTrans) store_w_tile (wReg, WsBuf[0], tid);
    else         store_w_tile_t(wReg, WsBuf[0], tid);
    __syncthreads();

    int cur = 0;
    for (int t = 0; t < nk; ++t) {
        // issue global loads for next tile (hidden behind WMMA below)
        const bool hasNext = (t + 1) < nk;
        if (hasNext) {
            int k0n = (t + 1) * BK;
            load_a_tile(A, lda, M, K, rowBase, k0n, tid, rowsIn, aReg);
            if (!wTrans) load_w_tile (W, ldw, N, K, colBase, k0n, tid, colsIn, wReg);
            else         load_w_tile_t(W, ldw, N, K, colBase, k0n, tid, colsIn, wReg);
        }

        // ---- compute on current buffer ----
        const _Float16* As = AsBuf[cur];
        const _Float16* Wc = WsBuf[cur];
        const _Float16* ap = &As[(wv * 16 + r) * LDT];
        v16h af;
        #pragma unroll
        for (int j = 0; j < 8; ++j) af[j]     = ap[8 * hf + j];        // K = 8h..8h+7
        #pragma unroll
        for (int j = 0; j < 8; ++j) af[8 + j] = ap[16 + 8 * hf + j];   // K = 16+8h..

        #pragma unroll
        for (int nt = 0; nt < 4; ++nt) {
            const _Float16* bp = &Wc[(nt * 16 + r) * LDT];
            v16h bf;
            #pragma unroll
            for (int j = 0; j < 8; ++j) bf[j]     = bp[8 * hf + j];
            #pragma unroll
            for (int j = 0; j < 8; ++j) bf[8 + j] = bp[16 + 8 * hf + j];
            acc[nt] = __builtin_amdgcn_wmma_f32_16x16x32_f16(
                false, af, false, bf, (short)0, acc[nt], false, false);
        }

        // write next tile into the other buffer (nobody reads it this iteration)
        if (hasNext) {
            store_a_tile(aReg, AsBuf[cur ^ 1], tid);
            if (!wTrans) store_w_tile (wReg, WsBuf[cur ^ 1], tid);
            else         store_w_tile_t(wReg, WsBuf[cur ^ 1], tid);
        }
        __syncthreads();
        cur ^= 1;
    }

    // ---- epilogue: C/D layout lane->(N = lane&15), vgpr i -> M = 8*(lane>>4)+i ----
    #pragma unroll
    for (int nt = 0; nt < 4; ++nt) {
        int col = colBase + nt * 16 + (lane & 15);
        if (col < N) {
            float bv = bias ? bias[col] : 0.0f;
            #pragma unroll
            for (int i = 0; i < 8; ++i) {
                int row = rowBase + wv * 16 + 8 * hf + i;
                if (row < M) {
                    float v = acc[nt][i] * scale + bv;
                    if (resid) v += resid[(size_t)row * ldc + col];
                    if (act == 1) v = gelu_exact(v);
                    C[(size_t)row * ldc + col] = v;
                }
            }
        }
    }
}

// ---------------- im2col for the stride-16 patch conv ----------------
__global__ __launch_bounds__(256)
void patch_gather_kernel(const float* __restrict__ x, float* __restrict__ patch)
{
    size_t idx = (size_t)blockIdx.x * 256 + threadIdx.x;
    size_t total = (size_t)cB * cNP * cPK;
    if (idx >= total) return;
    int k = (int)(idx % cPK);
    size_t pidx = idx / cPK;
    int b = (int)(pidx / cNP);
    int p = (int)(pidx % cNP);
    int ip = p / 14, jp = p % 14;
    int c = k / 256, rem = k % 256, pp = rem / 16, qq = rem % 16;
    patch[idx] = x[(((size_t)b * 3 + c) * 224 + (ip * 16 + pp)) * 224 + (jp * 16 + qq)];
}

// h[b,0,:] = cls + pos[0];  h[b,1+p,:] = tok[b*196+p] + pos[1+p]
__global__ __launch_bounds__(256)
void embed_finalize_kernel(const float* __restrict__ tok, const float* __restrict__ cls,
                           const float* __restrict__ pos, float* __restrict__ h)
{
    size_t idx = (size_t)blockIdx.x * 256 + threadIdx.x;
    size_t total = (size_t)cB * cSEQ * cD;
    if (idx >= total) return;
    int d = (int)(idx % cD);
    int s = (int)((idx / cD) % cSEQ);
    int b = (int)(idx / ((size_t)cD * cSEQ));
    float pe = pos[(size_t)s * cD + d];
    float v  = (s == 0) ? cls[d] : tok[((size_t)b * cNP + (s - 1)) * cD + d];
    h[idx] = v + pe;
}

// wave-per-row LayerNorm over 512 columns
__global__ __launch_bounds__(256)
void ln_kernel(const float* __restrict__ X, const float* __restrict__ gs,
               const float* __restrict__ gb, float* __restrict__ Y, int R)
{
    int row  = blockIdx.x * 8 + (threadIdx.x >> 5);
    int lane = threadIdx.x & 31;
    if (row >= R) return;
    const float* x = X + (size_t)row * cD;
    float s = 0.0f, sq = 0.0f;
    for (int i = lane; i < cD; i += 32) { float v = x[i]; s += v; sq += v * v; }
    #pragma unroll
    for (int o = 16; o; o >>= 1) { s += __shfl_xor(s, o, 32); sq += __shfl_xor(sq, o, 32); }
    float mean = s * (1.0f / cD);
    float var  = sq * (1.0f / cD) - mean * mean;
    float rstd = rsqrtf(var + cEPS);
    float* y = Y + (size_t)row * cD;
    for (int i = lane; i < cD; i += 32)
        y[i] = (x[i] - mean) * rstd * gs[i] + gb[i];
}

// wave-per-row softmax over 197 valid columns (row stride cSP)
__global__ __launch_bounds__(256)
void softmax_kernel(float* __restrict__ scores, int R)
{
    int row  = blockIdx.x * 8 + (threadIdx.x >> 5);
    int lane = threadIdx.x & 31;
    if (row >= R) return;
    float* p = scores + (size_t)row * cSP;
    float mx = -3.4e38f;
    for (int i = lane; i < cSEQ; i += 32) mx = fmaxf(mx, p[i]);
    #pragma unroll
    for (int o = 16; o; o >>= 1) mx = fmaxf(mx, __shfl_xor(mx, o, 32));
    float s = 0.0f;
    for (int i = lane; i < cSEQ; i += 32) { float e = __expf(p[i] - mx); p[i] = e; s += e; }
    #pragma unroll
    for (int o = 16; o; o >>= 1) s += __shfl_xor(s, o, 32);
    float inv = 1.0f / s;
    for (int i = lane; i < cSEQ; i += 32) p[i] *= inv;
}

// final LN on CLS row + 2-way classifier; one block per batch element
__global__ __launch_bounds__(256)
void head_kernel(const float* __restrict__ h, const float* __restrict__ gs,
                 const float* __restrict__ gb, const float* __restrict__ cw,
                 const float* __restrict__ cb, float* __restrict__ out)
{
    __shared__ float sh[cD];
    __shared__ float red[256];
    int b = blockIdx.x, tid = threadIdx.x;
    const float* row = h + (size_t)b * cSEQ * cD;   // CLS token row
    float s = 0.0f, sq = 0.0f;
    for (int d = tid; d < cD; d += 256) { float v = row[d]; sh[d] = v; s += v; sq += v * v; }
    red[tid] = s; __syncthreads();
    for (int o = 128; o; o >>= 1) { if (tid < o) red[tid] += red[tid + o]; __syncthreads(); }
    float mean = red[0] * (1.0f / cD); __syncthreads();
    red[tid] = sq; __syncthreads();
    for (int o = 128; o; o >>= 1) { if (tid < o) red[tid] += red[tid + o]; __syncthreads(); }
    float var  = red[0] * (1.0f / cD) - mean * mean;
    float rstd = rsqrtf(var + cEPS); __syncthreads();
    for (int d = tid; d < cD; d += 256) sh[d] = (sh[d] - mean) * rstd * gs[d] + gb[d];
    __syncthreads();
    for (int c = 0; c < 2; ++c) {
        float p = 0.0f;
        for (int d = tid; d < cD; d += 256) p += sh[d] * cw[(size_t)c * cD + d];
        red[tid] = p; __syncthreads();
        for (int o = 128; o; o >>= 1) { if (tid < o) red[tid] += red[tid + o]; __syncthreads(); }
        if (tid == 0) out[b * 2 + c] = red[0] + cb[c];
        __syncthreads();
    }
}

// ---------------- host orchestration ----------------
extern "C" void kernel_launch(void* const* d_in, const int* in_sizes, int n_in,
                              void* d_out, int out_size, void* d_ws, size_t ws_size,
                              hipStream_t stream)
{
    const float* x       = (const float*)d_in[0];
    const float* conv_w  = (const float*)d_in[1];
    const float* conv_b  = (const float*)d_in[2];
    const float* cls_tok = (const float*)d_in[3];
    const float* pos     = (const float*)d_in[4];
    const float* ln1_s   = (const float*)d_in[5];
    const float* ln1_b   = (const float*)d_in[6];
    const float* wq_w    = (const float*)d_in[7];
    const float* wq_b    = (const float*)d_in[8];
    const float* wk_w    = (const float*)d_in[9];
    const float* wk_b    = (const float*)d_in[10];
    const float* wv_w    = (const float*)d_in[11];
    const float* wv_b    = (const float*)d_in[12];
    const float* wo_w    = (const float*)d_in[13];
    const float* wo_b    = (const float*)d_in[14];
    const float* ln2_s   = (const float*)d_in[15];
    const float* ln2_b   = (const float*)d_in[16];
    const float* ff1_w   = (const float*)d_in[17];
    const float* ff1_b   = (const float*)d_in[18];
    const float* ff2_w   = (const float*)d_in[19];
    const float* ff2_b   = (const float*)d_in[20];
    const float* lnf_s   = (const float*)d_in[21];
    const float* lnf_b   = (const float*)d_in[22];
    const float* cls_w   = (const float*)d_in[23];
    const float* cls_b   = (const float*)d_in[24];
    (void)in_sizes; (void)n_in; (void)out_size;

    float* ws = (float*)d_ws;
    const size_t szH  = (size_t)cB * cSEQ * cD;        // 6,455,296
    const size_t szSc = (size_t)cB * cH * cSEQ * cSP;  // 20,979,712
    const size_t szFF = (size_t)cB * cSEQ * cFF;       // 25,821,184
    float* hbuf   = ws;
    float* ybuf   = hbuf + szH;
    float* qb     = ybuf + szH;
    float* kb     = qb + szH;
    float* vb     = kb + szH;
    float* cx     = vb + szH;
    float* scores = cx + szH;
    float* ffb    = scores + szSc;
    float* patch  = ffb;                               // alias (dead before layers)
    float* tok    = ffb + (size_t)cB * cNP * cPK;
    const size_t need = (6 * szH + szSc + szFF) * sizeof(float);
    if (ws_size < need) return;

    const int BS = cB * cSEQ;  // 12608 rows

    auto gemm = [&](const float* A, size_t aSB, size_t aSH, int lda,
                    const float* W, size_t wSB, size_t wSH, int ldw, int wT,
                    const float* bias, const float* resid,
                    float* C, size_t cSB, size_t cSH, int ldc,
                    int M, int N, int K, int Z, int batchH, float scale, int act) {
        dim3 g((unsigned)((N + BN - 1) / BN), (unsigned)((M + BM - 1) / BM), (unsigned)Z);
        gemm_wmma_kernel<<<g, 256, 0, stream>>>(A, aSB, aSH, lda, W, wSB, wSH, ldw, wT,
                                                bias, resid, C, cSB, cSH, ldc,
                                                M, N, K, batchH, scale, act);
    };

    // ---- patch embed ----
    {
        size_t tot = (size_t)cB * cNP * cPK;
        patch_gather_kernel<<<(unsigned)((tot + 255) / 256), 256, 0, stream>>>(x, patch);
        gemm(patch, 0, 0, cPK, conv_w, 0, 0, cPK, 0, conv_b, nullptr,
             tok, 0, 0, cD, cB * cNP, cD, cPK, 1, 1, 1.0f, 0);
        size_t tot2 = szH;
        embed_finalize_kernel<<<(unsigned)((tot2 + 255) / 256), 256, 0, stream>>>(tok, cls_tok, pos, hbuf);
    }

    // ---- transformer layers ----
    for (int l = 0; l < cL; ++l) {
        const size_t oD  = (size_t)l * cD;
        const size_t oDD = (size_t)l * cD * cD;
        const size_t oFD = (size_t)l * cFF * cD;

        ln_kernel<<<(unsigned)((BS + 7) / 8), 256, 0, stream>>>(hbuf, ln1_s + oD, ln1_b + oD, ybuf, BS);

        gemm(ybuf, 0, 0, cD, wq_w + oDD, 0, 0, cD, 0, wq_b + oD, nullptr, qb, 0, 0, cD, BS, cD, cD, 1, 1, 1.0f, 0);
        gemm(ybuf, 0, 0, cD, wk_w + oDD, 0, 0, cD, 0, wk_b + oD, nullptr, kb, 0, 0, cD, BS, cD, cD, 1, 1, 1.0f, 0);
        gemm(ybuf, 0, 0, cD, wv_w + oDD, 0, 0, cD, 0, wv_b + oD, nullptr, vb, 0, 0, cD, BS, cD, cD, 1, 1, 1.0f, 0);

        // scores[b,h,q,k] = (q . k) / 8  — batched over B*H
        gemm(qb, (size_t)cSEQ * cD, cDH, cD,
             kb, (size_t)cSEQ * cD, cDH, cD, 0,
             nullptr, nullptr,
             scores, (size_t)cH * cSEQ * cSP, (size_t)cSEQ * cSP, cSP,
             cSEQ, cSEQ, cDH, cB * cH, cH, 0.125f, 0);

        softmax_kernel<<<(unsigned)((cB * cH * cSEQ + 7) / 8), 256, 0, stream>>>(scores, cB * cH * cSEQ);

        // ctx[b,s,h*64+n] = attn @ v   (V staged as [K,N] via wTrans)
        gemm(scores, (size_t)cH * cSEQ * cSP, (size_t)cSEQ * cSP, cSP,
             vb, (size_t)cSEQ * cD, cDH, cD, 1,
             nullptr, nullptr,
             cx, (size_t)cSEQ * cD, cDH, cD,
             cSEQ, cDH, cSEQ, cB * cH, cH, 1.0f, 0);

        // out proj + residual
        gemm(cx, 0, 0, cD, wo_w + oDD, 0, 0, cD, 0, wo_b + oD, hbuf, hbuf, 0, 0, cD, BS, cD, cD, 1, 1, 1.0f, 0);

        ln_kernel<<<(unsigned)((BS + 7) / 8), 256, 0, stream>>>(hbuf, ln2_s + oD, ln2_b + oD, ybuf, BS);

        // FFN
        gemm(ybuf, 0, 0, cD, ff1_w + oFD, 0, 0, cD, 0, ff1_b + (size_t)l * cFF, nullptr,
             ffb, 0, 0, cFF, BS, cFF, cD, 1, 1, 1.0f, 1 /*gelu*/);
        gemm(ffb, 0, 0, cFF, ff2_w + oFD, 0, 0, cFF, 0, ff2_b + oD, hbuf,
             hbuf, 0, 0, cD, BS, cD, cFF, 1, 1, 1.0f, 0);
    }

    // ---- head ----
    head_kernel<<<(unsigned)cB, 256, 0, stream>>>(hbuf, lnf_s, lnf_b, cls_w, cls_b, (float*)d_out);
}
